// HGTransformerLayer_90177133347075
// MI455X (gfx1250) — matrix operand order repaired and checked by reading, not compiled, not run on hardware
//
#include <hip/hip_runtime.h>

// HGT layer for MI455X (gfx1250): algebraically-restructured attention,
// all GEMMs via v_wmma_f32_16x16x32_bf16 (wave32 WMMA, fp32 accum).
//
// BT=BN=1024, D=256, H=8, DH=32.

#define BTq 1024
#define BNq 1024
#define DD  256
#define NH  8
#define DHq 32

typedef __attribute__((ext_vector_type(16))) __bf16          v16bf;
typedef __attribute__((ext_vector_type(16))) unsigned short  v16us;
typedef __attribute__((ext_vector_type(8)))  float           v8f;

union FragU { v16us u; v16bf b; };

__device__ __forceinline__ unsigned short f2bf(float f) {
  unsigned int u = __float_as_uint(f);
  u += 0x7fffu + ((u >> 16) & 1u);          // round-to-nearest-even
  return (unsigned short)(u >> 16);
}
__device__ __forceinline__ float bf2f(unsigned short h) {
  return __uint_as_float(((unsigned int)h) << 16);
}

// ---- WMMA fragment loaders (CDNA5 ISA 7.12.2 layouts, wave32) ----
// A: 16x32 (MxK) bf16; element (m,k) at p[m*lda + k]. Lane L holds row M=L%16.
// Lanes 0-15: VGPR0-3 -> K=0..7, VGPR4-7 -> K=16..23; lanes 16-31: K=8..15 / 24..31.
__device__ __forceinline__ v16bf load_fragA(const unsigned short* p, int lda) {
  int lane = threadIdx.x & 31;
  int m = lane & 15, hi = lane >> 4;
  const unsigned short* row = p + m * lda;
  FragU f;
#pragma unroll
  for (int v = 0; v < 8; ++v) {
    int k0 = (v < 4) ? (8 * hi + 2 * v) : (16 + 8 * hi + 2 * (v - 4));
    unsigned int d = *(const unsigned int*)(row + k0);   // bf16 pair, K contiguous
    f.u[2 * v]     = (unsigned short)(d & 0xffffu);
    f.u[2 * v + 1] = (unsigned short)(d >> 16);
  }
  return f.b;
}

// B: 32x16 (KxN) bf16; element (k,n) at p[n*ldb + k] (column n is a K-contiguous row).
// Lane L holds column N=L%16; lanes 0-15 K=0..7,16..23, lanes 16-31 K=8..15,24..31.
__device__ __forceinline__ v16bf load_fragB(const unsigned short* p, int ldb) {
  int lane = threadIdx.x & 31;
  int n = lane & 15, hi = lane >> 4;
  const unsigned short* row = p + n * ldb;
  FragU f;
#pragma unroll
  for (int v = 0; v < 8; ++v) {
    int k0 = (v < 4) ? (8 * hi + 2 * v) : (16 + 8 * hi + 2 * (v - 4));
    unsigned int d = *(const unsigned int*)(row + k0);
    f.u[2 * v]     = (unsigned short)(d & 0xffffu);
    f.u[2 * v + 1] = (unsigned short)(d >> 16);
  }
  return f.b;
}

__device__ __forceinline__ v8f wmma_bf16(v16bf a, v16bf b, v8f c) {
  return __builtin_amdgcn_wmma_f32_16x16x32_bf16(
      /*neg_a=*/false, a, /*neg_b=*/false, b,
      /*c_mod=*/(short)0, c, /*reuse_a=*/false, /*reuse_b=*/false);
}

// ---------------- fp32 -> bf16 conversion ----------------
__global__ void cvt_bf16_kernel(const float* __restrict__ in,
                                unsigned short* __restrict__ out, int n) {
  int i = blockIdx.x * 256 + threadIdx.x;
  if (i < n) out[i] = f2bf(in[i]);
}

// ek = E @ Wk^T, ev = E @ Wv^T   (tiny: [2,256] each, fp32)
__global__ void ekev_kernel(const float* __restrict__ E,
                            const float* __restrict__ Wk,
                            const float* __restrict__ Wv,
                            float* __restrict__ ek, float* __restrict__ ev) {
  int tid = blockIdx.x * 256 + threadIdx.x;   // 0..511
  int t = tid >> 8, d = tid & 255;
  const float* e = E + t * DD;
  const float* wk = Wk + d * DD;
  const float* wv = Wv + d * DD;
  float sk = 0.f, sv = 0.f;
  for (int k = 0; k < DD; ++k) { sk += e[k] * wk[k]; sv += e[k] * wv[k]; }
  ek[t * DD + d] = sk;
  ev[t * DD + d] = sv;
}

// ---------------- projection GEMM: Y = (X @ W^T + bias) * scale -> bf16 ----------------
// A: bf16 [M][K], W: bf16 [N][K] (so B(k,n)=W[n][k], K contiguous).
// transposeOut=0: out[M][N] row-major; =1: out[N][M] (used for V -> V0T).
__global__ void proj_gemm_kernel(const unsigned short* __restrict__ A,
                                 const unsigned short* __restrict__ W,
                                 const float* __restrict__ bias,
                                 unsigned short* __restrict__ outb,
                                 int M, int N, int K, float scale, int transposeOut) {
  int tile = blockIdx.x * blockDim.y + threadIdx.y;   // one 16x16 tile per wave
  int tilesN = N >> 4;
  int tm = tile / tilesN, tn = tile % tilesN;
  if (tm * 16 >= M) return;
  int lane = threadIdx.x & 31;
  int nl = lane & 15, hi = lane >> 4;
  v8f c = {};
  const unsigned short* Abase = A + (size_t)tm * 16 * K;
  const unsigned short* Bbase = W + (size_t)tn * 16 * K;
  for (int kb = 0; kb < K; kb += 32) {
    v16bf a = load_fragA(Abase + kb, K);
    v16bf b = load_fragB(Bbase + kb, K);
    c = wmma_bf16(a, b, c);
  }
#pragma unroll
  for (int r = 0; r < 8; ++r) {
    int m = r + 8 * hi;
    int row = tm * 16 + m, col = tn * 16 + nl;
    float v = (c[r] + bias[col]) * scale;
    if (!transposeOut) outb[(size_t)row * N + col] = f2bf(v);
    else               outb[(size_t)col * M + row] = f2bf(v);
  }
}

// ---------------- fused attention: scores + masked softmax + AV + edge-correction ----------------
// One wave per (16-row target tile, head). Score strip [16][1024] kept in LDS (fp32),
// attn probabilities written back to LDS as bf16 for the AV WMMA pass.
__global__ void attn_kernel(const unsigned short* __restrict__ Qb,    // [1024][256] bf16 (scaled by 1/sqrt(DH))
                            const unsigned short* __restrict__ K0b,   // [1024][256] bf16
                            const unsigned short* __restrict__ V0Tb,  // [256][1024] bf16 (transposed V0)
                            const int* __restrict__ adj,              // [1024][1024]
                            const int* __restrict__ ety,              // [1024][1024]
                            const float* __restrict__ ek,             // [2][256]
                            const float* __restrict__ ev,             // [2][256]
                            unsigned short* __restrict__ outPre) {    // [1024][256] bf16
  extern __shared__ char smem[];
  float*          S   = (float*)smem;                               // 16*1024 f32 = 64KB
  unsigned short* P   = (unsigned short*)(smem + 16 * 1024 * 4);    // 16*1024 bf16 = 32KB
  float*          aux = (float*)(smem + 16 * 1024 * 4 + 16 * 1024 * 2); // qe0[16] qe1[16] ws0[16] ws1[16]

  const int h = blockIdx.y;
  const int b0 = blockIdx.x * 16;
  const int lane = threadIdx.x & 31;
  const int nl = lane & 15, hi = lane >> 4;

  __builtin_prefetch(adj + (size_t)b0 * BNq, 0, 0);
  __builtin_prefetch(ety + (size_t)b0 * BNq, 0, 0);

  // ---- phase 1: S[m][n] = Qh(b0+m) . K0h(n)  (Q prescaled) ----
  v16bf aq = load_fragA(Qb + (size_t)b0 * DD + h * DHq, DD);  // K = DH = 32: single WMMA per tile
  for (int nt = 0; nt < BNq / 16; ++nt) {
    v16bf bk = load_fragB(K0b + (size_t)nt * 16 * DD + h * DHq, DD);
    v8f c = {};
    c = wmma_bf16(aq, bk, c);
#pragma unroll
    for (int r = 0; r < 8; ++r)
      S[(r + 8 * hi) * BNq + nt * 16 + nl] = c[r];
  }

  // ---- qe[t] per row (q already carries the 1/sqrt(DH) scale) ----
  if (lane < 16) {
    int m = lane;
    float q0 = 0.f, q1 = 0.f;
    for (int kk = 0; kk < DHq; ++kk) {
      float q = bf2f(Qb[(size_t)(b0 + m) * DD + h * DHq + kk]);
      q0 += q * ek[h * DHq + kk];
      q1 += q * ek[DD + h * DHq + kk];
    }
    aux[m] = q0; aux[16 + m] = q1;
  }
  __syncthreads();

  // ---- phase 2: masked softmax per row; track edge-type attn sums ----
  for (int m = 0; m < 16; ++m) {
    const int* arow = adj + (size_t)(b0 + m) * BNq;
    const int* trow = ety + (size_t)(b0 + m) * BNq;
    float qe0 = aux[m], qe1 = aux[16 + m];
    float mx = -3.0e38f;
    for (int n = lane; n < BNq; n += 32) {
      float s = S[m * BNq + n];
      s = arow[n] ? (s + (trow[n] ? qe1 : qe0)) : -1.0e9f;
      S[m * BNq + n] = s;
      mx = fmaxf(mx, s);
    }
#pragma unroll
    for (int off = 16; off > 0; off >>= 1) mx = fmaxf(mx, __shfl_xor(mx, off, 32));
    float sAll = 0.f, sV = 0.f, s1 = 0.f;
    for (int n = lane; n < BNq; n += 32) {
      float e = __expf(S[m * BNq + n] - mx);
      S[m * BNq + n] = e;
      sAll += e;
      if (arow[n]) { sV += e; if (trow[n]) s1 += e; }
    }
#pragma unroll
    for (int off = 16; off > 0; off >>= 1) {
      sAll += __shfl_xor(sAll, off, 32);
      sV   += __shfl_xor(sV,   off, 32);
      s1   += __shfl_xor(s1,   off, 32);
    }
    float inv = sAll > 0.f ? 1.f / sAll : 0.f;
    for (int n = lane; n < BNq; n += 32) {
      float p = arow[n] ? S[m * BNq + n] * inv : 0.f;  // zero masked like reference
      P[m * BNq + n] = f2bf(p);
    }
    if (lane == 0) { aux[32 + m] = (sV - s1) * inv; aux[48 + m] = s1 * inv; }
  }
  __syncthreads();

  // ---- phase 3: O = attn @ V0_h  (K=1024 in chunks of 32), + edge-value correction ----
  v8f c0 = {}, c1 = {};
  for (int kt = 0; kt < BNq / 32; ++kt) {
    v16bf ap = load_fragA(P + kt * 32, BNq);
    v16bf b0f = load_fragB(V0Tb + (size_t)(h * DHq + 0)  * BNq + kt * 32, BNq);
    v16bf b1f = load_fragB(V0Tb + (size_t)(h * DHq + 16) * BNq + kt * 32, BNq);
    c0 = wmma_bf16(ap, b0f, c0);
    c1 = wmma_bf16(ap, b1f, c1);
  }
#pragma unroll
  for (int r = 0; r < 8; ++r) {
    int m = r + 8 * hi;
    int b = b0 + m;
    float w0 = aux[32 + m], w1 = aux[48 + m];
    int d0 = h * DHq + nl, d1 = h * DHq + 16 + nl;
    float o0 = c0[r] + w0 * ev[d0] + w1 * ev[DD + d0];
    float o1 = c1[r] + w0 * ev[d1] + w1 * ev[DD + d1];
    outPre[(size_t)b * DD + d0] = f2bf(o0);
    outPre[(size_t)b * DD + d1] = f2bf(o1);
  }
}

// ---------------- final GEMM: d_out = outPre @ Wo^T + bo (fp32 out) ----------------
__global__ void out_gemm_kernel(const unsigned short* __restrict__ A,
                                const unsigned short* __restrict__ W,
                                const float* __restrict__ bias,
                                float* __restrict__ out, int M, int N, int K) {
  int tile = blockIdx.x * blockDim.y + threadIdx.y;
  int tilesN = N >> 4;
  int tm = tile / tilesN, tn = tile % tilesN;
  if (tm * 16 >= M) return;
  int lane = threadIdx.x & 31;
  int nl = lane & 15, hi = lane >> 4;
  v8f c = {};
  const unsigned short* Abase = A + (size_t)tm * 16 * K;
  const unsigned short* Bbase = W + (size_t)tn * 16 * K;
  for (int kb = 0; kb < K; kb += 32) {
    v16bf a = load_fragA(Abase + kb, K);
    v16bf b = load_fragB(Bbase + kb, K);
    c = wmma_bf16(a, b, c);
  }
#pragma unroll
  for (int r = 0; r < 8; ++r) {
    int row = tm * 16 + r + 8 * hi, col = tn * 16 + nl;
    out[(size_t)row * N + col] = c[r] + bias[col];
  }
}

extern "C" void kernel_launch(void* const* d_in, const int* in_sizes, int n_in,
                              void* d_out, int out_size, void* d_ws, size_t ws_size,
                              hipStream_t stream) {
  (void)in_sizes; (void)n_in; (void)out_size; (void)ws_size;
  const float* hT  = (const float*)d_in[0];
  const float* hN  = (const float*)d_in[1];
  const int*   adj = (const int*)d_in[2];
  const int*   ety = (const int*)d_in[3];
  const float* Wq  = (const float*)d_in[4];
  const float* bq  = (const float*)d_in[5];
  const float* Wk  = (const float*)d_in[6];
  const float* bk  = (const float*)d_in[7];
  const float* Wv  = (const float*)d_in[8];
  const float* bv  = (const float*)d_in[9];
  const float* Wo  = (const float*)d_in[10];
  const float* bo  = (const float*)d_in[11];
  const float* E   = (const float*)d_in[12];
  float* out = (float*)d_out;

  // workspace carve (256B aligned), total ~3.7MB
  char* w = (char*)d_ws;
  size_t off = 0;
  auto alloc = [&](size_t bytes) -> char* {
    off = (off + 255) & ~(size_t)255;
    char* p = w + off; off += bytes; return p;
  };
  unsigned short* hTb   = (unsigned short*)alloc((size_t)BTq * DD * 2);
  unsigned short* hNb   = (unsigned short*)alloc((size_t)BNq * DD * 2);
  unsigned short* Wqb   = (unsigned short*)alloc((size_t)DD * DD * 2);
  unsigned short* Wkb   = (unsigned short*)alloc((size_t)DD * DD * 2);
  unsigned short* Wvb   = (unsigned short*)alloc((size_t)DD * DD * 2);
  unsigned short* Wob   = (unsigned short*)alloc((size_t)DD * DD * 2);
  unsigned short* Qb    = (unsigned short*)alloc((size_t)BTq * DD * 2);
  unsigned short* K0b   = (unsigned short*)alloc((size_t)BNq * DD * 2);
  unsigned short* V0Tb  = (unsigned short*)alloc((size_t)DD * BNq * 2);
  unsigned short* oPre  = (unsigned short*)alloc((size_t)BTq * DD * 2);
  float*          ekW   = (float*)alloc(2 * DD * 4);
  float*          evW   = (float*)alloc(2 * DD * 4);

  // 1) fp32 -> bf16 conversions
  cvt_bf16_kernel<<<(BTq * DD + 255) / 256, 256, 0, stream>>>(hT, hTb, BTq * DD);
  cvt_bf16_kernel<<<(BNq * DD + 255) / 256, 256, 0, stream>>>(hN, hNb, BNq * DD);
  cvt_bf16_kernel<<<(DD * DD + 255) / 256, 256, 0, stream>>>(Wq, Wqb, DD * DD);
  cvt_bf16_kernel<<<(DD * DD + 255) / 256, 256, 0, stream>>>(Wk, Wkb, DD * DD);
  cvt_bf16_kernel<<<(DD * DD + 255) / 256, 256, 0, stream>>>(Wv, Wvb, DD * DD);
  cvt_bf16_kernel<<<(DD * DD + 255) / 256, 256, 0, stream>>>(Wo, Wob, DD * DD);

  // 2) tiny edge-type projections ek = E@Wk^T, ev = E@Wv^T
  ekev_kernel<<<2, 256, 0, stream>>>(E, Wk, Wv, ekW, evW);

  // 3) projections (WMMA): Q (scaled by 1/sqrt(DH)), K0, V0 (stored transposed)
  const float qscale = 0.17677669529663687f;  // 1/sqrt(32)
  dim3 gblk(32, 4);
  int ptiles = (BTq / 16) * (DD / 16);        // 1024 tiles
  proj_gemm_kernel<<<ptiles / 4, gblk, 0, stream>>>(hTb, Wqb, bq, Qb,  BTq, DD, DD, qscale, 0);
  proj_gemm_kernel<<<ptiles / 4, gblk, 0, stream>>>(hNb, Wkb, bk, K0b, BNq, DD, DD, 1.0f,   0);
  proj_gemm_kernel<<<ptiles / 4, gblk, 0, stream>>>(hNb, Wvb, bv, V0Tb,BNq, DD, DD, 1.0f,   1);

  // 4) fused attention: 64 row-tiles x 8 heads, 1 wave each, 96.25KB dynamic LDS
  size_t lds = (size_t)16 * 1024 * 4 + (size_t)16 * 1024 * 2 + 64 * 4;
  attn_kernel<<<dim3(BTq / 16, NH), 32, lds, stream>>>(Qb, K0b, V0Tb, adj, ety, ekW, evW, oPre);

  // 5) output projection -> fp32 d_out
  out_gemm_kernel<<<ptiles / 4, gblk, 0, stream>>>(oPre, Wob, bo, out, BTq, DD, DD);
}
